// GATEncoder_29764123361883
// MI455X (gfx1250) — matrix operand order repaired and checked by reading, not compiled, not run on hardware
//
#include <hip/hip_runtime.h>

typedef __attribute__((ext_vector_type(16))) __bf16 v16bf;
typedef __attribute__((ext_vector_type(8)))  __bf16 v8bf;
typedef __attribute__((ext_vector_type(8)))  float  v8f;

__device__ inline __bf16 f2bf(float f) {
  unsigned u = __float_as_uint(f);
  unsigned r = u + 0x7FFFu + ((u >> 16) & 1u);   // round-to-nearest-even
  unsigned short hs = (unsigned short)(r >> 16);
  return __builtin_bit_cast(__bf16, hs);
}

// A fragment: 16x32 bf16 tile from LDS (contiguous k order, same as B).
// lane = half*16 + mrow ; element j -> K = half*16 + j
// => two contiguous 8-element (16B) runs per lane: ds_load_b128 x2.
__device__ inline v16bf load_a_lds(const __bf16* base, int ld, int mrow, int half, int k0) {
  const __bf16* p = base + mrow * ld + k0 + (half << 4);
  v8bf lo = *(const v8bf*)(p);
  v8bf hi = *(const v8bf*)(p + 8);
  return __builtin_shufflevector(lo, hi, 0, 1, 2, 3, 4, 5, 6, 7,
                                         8, 9, 10, 11, 12, 13, 14, 15);
}

// B fragment: B[k][n] = Wb[n][k], Wb bf16 row-major (pre-converted, L2-resident).
// lane j -> K = half*16 + j : 16 contiguous bf16 = one 32B read per lane.
__device__ inline v16bf load_b_bf(const __bf16* __restrict__ Wb, int ldw, int ncol,
                                  int half, int k0) {
  return *(const v16bf*)(Wb + (long)ncol * ldw + k0 + (half << 4));
}

__device__ inline v8f wmma_bf16(v16bf a, v16bf b, v8f c) {
  return __builtin_amdgcn_wmma_f32_16x16x32_bf16(false, a, false, b, (short)0, c, false, false);
}

// ---------------- Kernel 0: one-shot fp32 -> bf16 weight conversion ----------------
__global__ __launch_bounds__(256) void k_cvtw(
    const float* __restrict__ w1, const float* __restrict__ w2,
    const float* __restrict__ gw, const float* __restrict__ pw,
    __bf16* __restrict__ w1b, __bf16* __restrict__ w2b,
    __bf16* __restrict__ gwb, __bf16* __restrict__ pwb)
{
  int i = blockIdx.x * 256 + threadIdx.x;
  if (i < 512 * 256) w1b[i] = f2bf(w1[i]);
  if (i < 64 * 512)  w2b[i] = f2bf(w2[i]);
  if (i < 128 * 64)  gwb[i] = f2bf(gw[i]);
  if (i < 64 * 64)   pwb[i] = f2bf(pw[i]);
}

// ---------------- Kernel 1: fused pre_fc  z = relu(x@w1^T+b1)@w2^T+b2 ----------------
__global__ __launch_bounds__(128) void k_prefc(
    const float* __restrict__ x,
    const __bf16* __restrict__ w1b, const float* __restrict__ b1,
    const __bf16* __restrict__ w2b, const float* __restrict__ b2,
    float* __restrict__ z, int N)
{
  __shared__ __bf16 xs[16 * 256];   // 8 KB
  __shared__ __bf16 hid[16 * 512];  // 16 KB
  const int t = threadIdx.x;
  const int wave = t >> 5, lane = t & 31;
  const int mrow = lane & 15, half = lane >> 4;
  const long row0 = (long)blockIdx.x * 16;

  for (int i = t; i < 16 * 256; i += 128) {
    int m = i >> 8, k = i & 255;
    long r = row0 + m; if (r >= N) r = N - 1;
    xs[i] = f2bf(x[r * 256 + k]);
  }
  __syncthreads();

  // Stage 1: hidden[16,512]; wave owns cols [wave*128, wave*128+128)
  for (int tt = 0; tt < 8; ++tt) {
    int n0 = wave * 128 + tt * 16;
    v8f acc = {};
#pragma unroll
    for (int ks = 0; ks < 8; ++ks) {
      v16bf a = load_a_lds(xs, 256, mrow, half, ks * 32);
      v16bf b = load_b_bf(w1b, 256, n0 + mrow, half, ks * 32);
      acc = wmma_bf16(a, b, acc);
    }
    int n = n0 + mrow;
    float bias = b1[n];
#pragma unroll
    for (int r = 0; r < 8; ++r) {
      int m = (half << 3) + r;
      float v = acc[r] + bias;
      hid[m * 512 + n] = f2bf(v > 0.f ? v : 0.f);
    }
  }
  __syncthreads();

  // Stage 2: z[16,64]; wave owns cols [wave*16, wave*16+16)
  {
    int n0 = wave * 16;
    v8f acc = {};
#pragma unroll
    for (int ks = 0; ks < 16; ++ks) {
      v16bf a = load_a_lds(hid, 512, mrow, half, ks * 32);
      v16bf b = load_b_bf(w2b, 512, n0 + mrow, half, ks * 32);
      acc = wmma_bf16(a, b, acc);
    }
    int n = n0 + mrow;
    float bias = b2[n];
#pragma unroll
    for (int r = 0; r < 8; ++r) {
      int m = (half << 3) + r;
      long rr = row0 + m;
      if (rr < N) z[rr * 64 + n] = acc[r] + bias;
    }
  }
}

// ---------------- Kernel 2: h = z@gat_w^T ; a_src/a_dst attention dots ----------------
__global__ __launch_bounds__(128) void k_gath(
    const float* __restrict__ z, const __bf16* __restrict__ gwb,
    const float* __restrict__ att_src, const float* __restrict__ att_dst,
    float* __restrict__ h, float* __restrict__ a_src, float* __restrict__ a_dst, int N)
{
  __shared__ __bf16 zs[16 * 64];
  __shared__ float  hs[16 * 128];
  const int t = threadIdx.x;
  const int wave = t >> 5, lane = t & 31;
  const int mrow = lane & 15, half = lane >> 4;
  const long row0 = (long)blockIdx.x * 16;

  for (int i = t; i < 16 * 64; i += 128) {
    int m = i >> 6, k = i & 63;
    long r = row0 + m; if (r >= N) r = N - 1;
    zs[i] = f2bf(z[r * 64 + k]);
  }
  __syncthreads();

  for (int tt = 0; tt < 2; ++tt) {
    int n0 = (wave * 2 + tt) * 16;            // 8 tiles cover 128 cols
    v8f acc = {};
#pragma unroll
    for (int ks = 0; ks < 2; ++ks) {
      v16bf a = load_a_lds(zs, 64, mrow, half, ks * 32);
      v16bf b = load_b_bf(gwb, 64, n0 + mrow, half, ks * 32);
      acc = wmma_bf16(a, b, acc);
    }
    int n = n0 + mrow;
#pragma unroll
    for (int r = 0; r < 8; ++r) {
      int m = (half << 3) + r;
      hs[m * 128 + n] = acc[r];
      long rr = row0 + m;
      if (rr < N) h[rr * 128 + n] = acc[r];
    }
  }
  __syncthreads();

  if (t < 64) {                                // (m, head, src|dst) dot products
    int which = t >> 5;
    int c = t & 31;
    int m = c >> 1, head = c & 1;
    const float* att = which ? att_dst : att_src;
    float s = 0.f;
    for (int d = 0; d < 64; ++d) s += hs[m * 128 + head * 64 + d] * att[head * 64 + d];
    long r = row0 + m;
    if (r < N) (which ? a_dst : a_src)[r * 2 + head] = s;
  }
}

// ---------------- Kernel 3: init scratch ----------------
__global__ void k_init(int* __restrict__ amax, float* __restrict__ denom,
                       float* __restrict__ accum, int N)
{
  long i = (long)blockIdx.x * blockDim.x + threadIdx.x;
  if (i < (long)N * 64) accum[i] = 0.f;
  if (i < (long)N * 2) { denom[i] = 0.f; amax[i] = (int)0xFF800000u; } // -inf
}

__device__ inline void edge_sd(const int* __restrict__ e, long i, long E, int N,
                               int& s, int& d)
{
  if (i < E) { s = e[i]; d = e[E + i]; }
  else { s = (int)(i - E); d = s; }
  s = s < 0 ? 0 : (s >= N ? N - 1 : s);
  d = d < 0 ? 0 : (d >= N ? N - 1 : d);
}

// ---------------- Kernel 4: segment max of leaky_relu(a_s+a_d) ----------------
__global__ void k_amax(const int* __restrict__ e, const float* __restrict__ a_src,
                       const float* __restrict__ a_dst, int* __restrict__ amax,
                       long E, int N)
{
  long i = (long)blockIdx.x * blockDim.x + threadIdx.x;
  if (i >= E + N) return;
  int s, d; edge_sd(e, i, E, N, s, d);
#pragma unroll
  for (int hh = 0; hh < 2; ++hh) {
    float v = a_src[(long)s * 2 + hh] + a_dst[(long)d * 2 + hh];
    float al = v > 0.f ? v : 0.2f * v;
    int* addr = amax + (long)d * 2 + hh;
    if (al >= 0.f) atomicMax(addr, __float_as_int(al));
    else           atomicMin((unsigned int*)addr, __float_as_uint(al));
  }
}

// ---------------- Kernel 5: softmax denominators ----------------
__global__ void k_denom(const int* __restrict__ e, const float* __restrict__ a_src,
                        const float* __restrict__ a_dst, const int* __restrict__ amax,
                        float* __restrict__ denom, long E, int N)
{
  long i = (long)blockIdx.x * blockDim.x + threadIdx.x;
  if (i >= E + N) return;
  int s, d; edge_sd(e, i, E, N, s, d);
#pragma unroll
  for (int hh = 0; hh < 2; ++hh) {
    float v = a_src[(long)s * 2 + hh] + a_dst[(long)d * 2 + hh];
    float al = v > 0.f ? v : 0.2f * v;
    float am = __int_as_float(amax[(long)d * 2 + hh]);
    if (!(am > -1e38f && am < 1e38f)) am = 0.f;
    atomicAdd(denom + (long)d * 2 + hh, __expf(al - am));
  }
}

// ---------------- Kernel 6: weighted message scatter (head-mean folded in) ----------------
__global__ __launch_bounds__(256) void k_scatter(
    const int* __restrict__ e, const float* __restrict__ a_src, const float* __restrict__ a_dst,
    const int* __restrict__ amax, const float* __restrict__ denom,
    const float* __restrict__ h, float* __restrict__ accum, long E, int N)
{
  long edge = (long)blockIdx.x * 8 + (threadIdx.x >> 5);  // 8 edges / block, wave per edge
  int lane = threadIdx.x & 31;
  if (edge >= E + N) return;
  int s, d; edge_sd(e, edge, E, N, s, d);
  float w[2];
#pragma unroll
  for (int hh = 0; hh < 2; ++hh) {
    float v = a_src[(long)s * 2 + hh] + a_dst[(long)d * 2 + hh];
    float al = v > 0.f ? v : 0.2f * v;
    float am = __int_as_float(amax[(long)d * 2 + hh]);
    if (!(am > -1e38f && am < 1e38f)) am = 0.f;
    float ex = __expf(al - am);
    w[hh] = 0.5f * ex / (denom[(long)d * 2 + hh] + 1e-16f);   // 0.5 = head mean
  }
  const float* hsrc = h + (long)s * 128;
  float* adst = accum + (long)d * 64;
#pragma unroll
  for (int p = 0; p < 2; ++p) {
    int dd = lane + p * 32;
    float msg = w[0] * hsrc[dd] + w[1] * hsrc[64 + dd];
    atomicAdd(adst + dd, msg);
  }
}

// ---------------- Kernel 7: finalize + post GEMM ----------------
__global__ __launch_bounds__(128) void k_post(
    const float* __restrict__ accum, const float* __restrict__ resid,
    const float* __restrict__ gat_b, const __bf16* __restrict__ pwb,
    const float* __restrict__ post_b, float* __restrict__ out, int N)
{
  __shared__ __bf16 zs[16 * 64];
  const int t = threadIdx.x;
  const int wave = t >> 5, lane = t & 31;
  const int mrow = lane & 15, half = lane >> 4;
  const long row0 = (long)blockIdx.x * 16;

  for (int i = t; i < 16 * 64; i += 128) {
    int m = i >> 6, dc = i & 63;
    long r = row0 + m; if (r >= N) r = N - 1;
    float g = accum[r * 64 + dc] + gat_b[dc];
    float act = g > 0.f ? g : 0.01f * g;
    zs[i] = f2bf(act + resid[r * 64 + dc]);
  }
  __syncthreads();

  int n0 = wave * 16;   // 4 waves cover 64 output cols
  v8f acc = {};
#pragma unroll
  for (int ks = 0; ks < 2; ++ks) {
    v16bf a = load_a_lds(zs, 64, mrow, half, ks * 32);
    v16bf b = load_b_bf(pwb, 64, n0 + mrow, half, ks * 32);
    acc = wmma_bf16(a, b, acc);
  }
  int n = n0 + mrow;
  float bias = post_b[n];
#pragma unroll
  for (int r = 0; r < 8; ++r) {
    int m = (half << 3) + r;
    long rr = row0 + m;
    if (rr < N) out[rr * 64 + n] = acc[r] + bias;
  }
}

extern "C" void kernel_launch(void* const* d_in, const int* in_sizes, int n_in,
                              void* d_out, int out_size, void* d_ws, size_t ws_size,
                              hipStream_t stream)
{
  const float* x       = (const float*)d_in[0];
  const int*   e       = (const int*)  d_in[1];
  const float* w1      = (const float*)d_in[2];
  const float* b1      = (const float*)d_in[3];
  const float* w2      = (const float*)d_in[4];
  const float* b2      = (const float*)d_in[5];
  const float* gat_w   = (const float*)d_in[6];
  const float* att_src = (const float*)d_in[7];
  const float* att_dst = (const float*)d_in[8];
  const float* gat_b   = (const float*)d_in[9];
  const float* post_w  = (const float*)d_in[10];
  const float* post_b  = (const float*)d_in[11];
  float* out = (float*)d_out;

  const int  N = in_sizes[0] / 256;
  const long E = in_sizes[1] / 2;

  // fp32 workspace: z | h | a_src | a_dst | amax | denom | accum  = N*264 floats
  float* ws    = (float*)d_ws;
  float* z     = ws;                        // N*64  (also resid)
  float* h     = z     + (size_t)N * 64;    // N*128
  float* a_src = h     + (size_t)N * 128;   // N*2
  float* a_dst = a_src + (size_t)N * 2;     // N*2
  int*   amax  = (int*)(a_dst + (size_t)N * 2);      // N*2 (float bits)
  float* denom = (float*)amax + (size_t)N * 2;       // N*2
  float* accum = denom + (size_t)N * 2;     // N*64
  // bf16 weight copies (L2-resident, converted once per launch)
  __bf16* w1b = (__bf16*)(accum + (size_t)N * 64);   // 512*256
  __bf16* w2b = w1b + 512 * 256;                     // 64*512
  __bf16* gwb = w2b + 64 * 512;                      // 128*64
  __bf16* pwb = gwb + 128 * 64;                      // 64*64

  const int  nb16 = (N + 15) / 16;
  const long Et   = E + N;

  k_cvtw<<<dim3((512 * 256 + 255) / 256), dim3(256), 0, stream>>>(
      w1, w2, gat_w, post_w, w1b, w2b, gwb, pwb);
  k_init<<<dim3((unsigned)((N * 64 + 255) / 256)), dim3(256), 0, stream>>>(amax, denom, accum, N);
  k_prefc<<<dim3((unsigned)nb16), dim3(128), 0, stream>>>(x, w1b, b1, w2b, b2, z, N);
  k_gath<<<dim3((unsigned)nb16), dim3(128), 0, stream>>>(z, gwb, att_src, att_dst, h, a_src, a_dst, N);
  k_amax<<<dim3((unsigned)((Et + 255) / 256)), dim3(256), 0, stream>>>(e, a_src, a_dst, amax, E, N);
  k_denom<<<dim3((unsigned)((Et + 255) / 256)), dim3(256), 0, stream>>>(e, a_src, a_dst, amax, denom, E, N);
  k_scatter<<<dim3((unsigned)((Et + 7) / 8)), dim3(256), 0, stream>>>(e, a_src, a_dst, amax, denom, h, accum, E, N);
  k_post<<<dim3((unsigned)nb16), dim3(128), 0, stream>>>(accum, z, gat_b, pwb, post_b, out, N);
}
